// RNNBezierDecoder_72138270703916
// MI455X (gfx1250) — matrix operand order repaired and checked by reading, not compiled, not run on hardware
//
#include <hip/hip_runtime.h>
#include <hip/hip_bf16.h>
#include <stdint.h>

// Problem dims (fixed by the reference)
#define TT   320
#define BB   192
#define HH   1024
#define G4   4096      // 4*HH
#define LATD 24
#define NPTS 8

typedef __bf16 bf16_t;
typedef __attribute__((ext_vector_type(16))) __bf16 v16bf;
typedef __attribute__((ext_vector_type(8)))  float  v8f;

union AFrag { uint4 u[2]; v16bf v; };

// Native gfx1250 v_tanh_f32 when available (straight-line TRANS op instead of
// the branchy libm tanhf/expf expansions); fall back to tanhf otherwise.
#if __has_builtin(__builtin_amdgcn_tanhf)
__device__ __forceinline__ float ftanh(float x) { return __builtin_amdgcn_tanhf(x); }
#elif __has_builtin(__builtin_amdgcn_tanh_f32)
__device__ __forceinline__ float ftanh(float x) { return __builtin_amdgcn_tanh_f32(x); }
#else
__device__ __forceinline__ float ftanh(float x) { return tanhf(x); }
#endif
// sigmoid(x) = 0.5*tanh(x/2) + 0.5  -> also a single v_tanh
__device__ __forceinline__ float sigf(float v) { return 0.5f * ftanh(0.5f * v) + 0.5f; }

// Monotonic grid barrier over the 48 blocks of one direction.
__device__ __forceinline__ void gbar(unsigned* cnt, unsigned target) {
  __syncthreads();
  if (threadIdx.x == 0) {
    __threadfence();                       // release our h/c writes to device scope
    atomicAdd(cnt, 1u);
    while (atomicAdd(cnt, 0u) < target) __builtin_amdgcn_s_sleep(8);
  }
  __syncthreads();
  __threadfence();                         // acquire: invalidate stale L0 lines of h
}

// ---------------------------------------------------------------------------
// Setup: zero scratch, bf16-convert W_hh, compute tanh initial h0/c0 states.
// ---------------------------------------------------------------------------
__global__ void setup_kernel(const float* __restrict__ cpts, const float* __restrict__ ratw,
                             const float* __restrict__ W_hf, const float* __restrict__ b_hf,
                             const float* __restrict__ W_hb, const float* __restrict__ b_hb,
                             const float* __restrict__ W_cf, const float* __restrict__ b_cf,
                             const float* __restrict__ W_cb, const float* __restrict__ b_cb,
                             const float* __restrict__ Whh_f, const float* __restrict__ Whh_b,
                             bf16_t* __restrict__ Wbf, bf16_t* __restrict__ hping,
                             float* __restrict__ cbuf, float* __restrict__ logits,
                             unsigned* __restrict__ bar)
{
  const size_t i0 = (size_t)blockIdx.x * blockDim.x + threadIdx.x;
  const size_t st = (size_t)gridDim.x * blockDim.x;

  if (i0 < 2) bar[i0] = 0u;
  for (size_t i = i0; i < (size_t)BB * TT; i += st) logits[i] = 0.f;

  // W_hh (fp32) -> bf16, both directions, L2-resident thereafter
  const size_t NW = (size_t)G4 * HH;
  for (size_t i = i0; i < 2 * NW; i += st) {
    float v = (i < NW) ? Whh_f[i] : Whh_b[i - NW];
    Wbf[i] = (bf16_t)v;
  }

  // h0/c0 = tanh(latent @ W.T + b); h0 -> bf16 ping buffer phase 0, c0 -> f32
  for (size_t i = i0; i < (size_t)2 * BB * HH; i += st) {
    const int d   = (int)(i / (BB * HH));
    const int rem = (int)(i % (BB * HH));
    const int b   = rem / HH, j = rem % HH;
    const float* Wh = d ? W_hb : W_hf;  const float* bh = d ? b_hb : b_hf;
    const float* Wc = d ? W_cb : W_cf;  const float* bc = d ? b_cb : b_cf;
    float ah = bh[j], ac = bc[j];
    #pragma unroll
    for (int k = 0; k < LATD; ++k) {
      float lv = (k < 16) ? cpts[b * 16 + k] : ratw[b * 8 + (k - 16)];
      ah += lv * Wh[j * LATD + k];
      ac += lv * Wc[j * LATD + k];
    }
    hping[(size_t)(d * 2 + 0) * BB * HH + rem] = (bf16_t)ftanh(ah);  // phase 0
    cbuf [(size_t)d * BB * HH + rem]           = ftanh(ac);
  }
}

// ---------------------------------------------------------------------------
// Persistent bi-LSTM: 96 blocks (48 per direction), 8 waves each.
// Per step: g = h @ W_hh.T (bf16 WMMA, f32 acc) + x@W_ih.T + biases, gate math,
// h update (bf16 ping-pong), fused logits partial via atomicAdd.
// ---------------------------------------------------------------------------
__global__ __launch_bounds__(256)
void lstm_persistent_kernel(const float* __restrict__ x,
                            const float* __restrict__ Wih_f, const float* __restrict__ bih_f,
                            const float* __restrict__ bhh_f,
                            const float* __restrict__ Wih_b, const float* __restrict__ bih_b,
                            const float* __restrict__ bhh_b,
                            const float* __restrict__ Wlog,
                            const bf16_t* __restrict__ Wbf,
                            bf16_t* __restrict__ hping,
                            float* __restrict__ cbuf,
                            float* __restrict__ logits,
                            unsigned* __restrict__ bar)
{
  const int tid   = threadIdx.x;
  const int lane  = tid & 31;
  const int wave  = tid >> 5;
  const int nl    = lane & 15;     // N (B-frag/C-frag) or M (A-frag)
  const int khalf = lane >> 4;     // which K half the lane holds

  const int blk = blockIdx.x;
  const int dir = blk / 48;        // 0=forward, 1=backward
  const int loc = blk % 48;
  const int mt  = loc % 12;        // 16-row batch tile
  const int ncg = loc / 12;        // column group: 16 n-tiles

  const float*  Wih = dir ? Wih_b : Wih_f;
  const float*  bih = dir ? bih_b : bih_f;
  const float*  bhh = dir ? bhh_b : bhh_f;
  const bf16_t* W   = Wbf  + (size_t)dir * G4 * HH;
  float*        cp  = cbuf + (size_t)dir * BB * HH;
  unsigned*   mybar = bar + dir;

  int nt[2]; nt[0] = ncg * 16 + wave * 2; nt[1] = nt[0] + 1;

  // Step-invariant per-lane constants: input-proj weights, summed biases, W_log.
  float wih0[2][4], wih1[2][4], bsum[2][4], wlog[2];
  #pragma unroll
  for (int t2 = 0; t2 < 2; ++t2) {
    const int col = nt[t2] * 16 + nl;
    wlog[t2] = Wlog[dir * HH + col];
    #pragma unroll
    for (int g = 0; g < 4; ++g) {
      const int row = g * HH + col;
      wih0[t2][g] = Wih[row * 2 + 0];
      wih1[t2][g] = Wih[row * 2 + 1];
      bsum[t2][g] = bih[row] + bhh[row];
    }
  }
  // Runtime part of the B-fragment address (per lane); the per-(tile,gate)
  // deltas are compile-time constants folded into the load's 24-bit immediate.
  // B 32x16 bf16 layout: lane N = nl, 16 contiguous K at 16*khalf (row-major W_hh)
  const uint32_t bbase = (uint32_t)((((uint32_t)(nt[0] * 16 + nl)) * HH + khalf * 16)
                                    * sizeof(bf16_t));
  // A-fragment LDS byte offset: lane M = nl, 16 contiguous K at 16*khalf
  const uint32_t aoff  = (uint32_t)((nl * HH + khalf * 16) * sizeof(bf16_t));

  __shared__ bf16_t sA[16 * HH];   // 16 batch rows x 1024 K of h (32 KB)
  __shared__ float  sX[32];        // x[t] for this row tile

  #pragma unroll 1
  for (int s = 0; s < TT; ++s) {
    const bf16_t* hin  = hping + (size_t)(dir * 2 + (s & 1))       * BB * HH;
    bf16_t*       hout = hping + (size_t)(dir * 2 + ((s + 1) & 1)) * BB * HH;
    const int t = dir ? (TT - 1 - s) : s;

    // Cooperative stage of the A slab into LDS (shared by all 8 waves)
    {
      const uint4* src = (const uint4*)(hin + (size_t)mt * 16 * HH);
      uint4* dst = (uint4*)sA;
      #pragma unroll
      for (int i = 0; i < 8; ++i) dst[tid + 256 * i] = src[tid + 256 * i];
      if (tid < 16) {
        sX[tid * 2 + 0] = x[((size_t)t * BB + mt * 16 + tid) * 2 + 0];
        sX[tid * 2 + 1] = x[((size_t)t * BB + mt * 16 + tid) * 2 + 1];
      }
    }
    __syncthreads();

    const v8f vz = {0.f, 0.f, 0.f, 0.f, 0.f, 0.f, 0.f, 0.f};
    v8f acc[2][4];
    #pragma unroll
    for (int t2 = 0; t2 < 2; ++t2)
      #pragma unroll
      for (int g = 0; g < 4; ++g) acc[t2][g] = vz;

    // K loop: 32 chunks of K=32. A from LDS; B streams from L2-hot bf16 W_hh.
    // unroll 2: 16 WMMAs in flight for deeper load/WMMA overlap, still no spill.
    #pragma unroll 2
    for (int kb = 0; kb < 32; ++kb) {
      const uint32_t ckb = (uint32_t)kb * 64u;   // 32 K-elements * 2 bytes
      AFrag a;
      a.u[0] = *(const uint4*)((const char*)sA + (aoff + ckb));
      a.u[1] = *(const uint4*)((const char*)sA + (aoff + ckb) + 16u);
      const char* wb = (const char*)W + (bbase + ckb);
      #pragma unroll
      for (int t2 = 0; t2 < 2; ++t2) {
        #pragma unroll
        for (int g = 0; g < 4; ++g) {
          // constant delta -> folded into the global_load immediate offset
          const v16bf bfr = *(const v16bf*)(wb + (size_t)(t2 * 16 * HH + g * HH * HH)
                                                 * sizeof(bf16_t));
          acc[t2][g] = __builtin_amdgcn_wmma_f32_16x16x32_bf16(
              false, a.v, false, bfr, (short)0, acc[t2][g], false, false);
        }
      }
    }

    // Epilogue: gates, c/h update, fused logits partial (all v_tanh based).
    #pragma unroll
    for (int t2 = 0; t2 < 2; ++t2) {
      const int ng = nt[t2] * 16 + nl;
      #pragma unroll
      for (int r = 0; r < 8; ++r) {
        const int  bg = mt * 16 + r + 8 * khalf;  // C layout: VGPR r -> M=r / M=8+r
        const float x0 = sX[(r + 8 * khalf) * 2 + 0];
        const float x1 = sX[(r + 8 * khalf) * 2 + 1];
        const float pi = acc[t2][0][r] + bsum[t2][0] + x0 * wih0[t2][0] + x1 * wih1[t2][0];
        const float pf = acc[t2][1][r] + bsum[t2][1] + x0 * wih0[t2][1] + x1 * wih1[t2][1];
        const float pg = acc[t2][2][r] + bsum[t2][2] + x0 * wih0[t2][2] + x1 * wih1[t2][2];
        const float po = acc[t2][3][r] + bsum[t2][3] + x0 * wih0[t2][3] + x1 * wih1[t2][3];
        const size_t ci = (size_t)bg * HH + ng;
        const float cold = cp[ci];
        const float cnew = sigf(pf) * cold + sigf(pi) * ftanh(pg);
        cp[ci] = cnew;
        const float hval = sigf(po) * ftanh(cnew);
        hout[ci] = (bf16_t)hval;
        // hs is never stored: reduce h * W_log across the 16 lanes sharing row bg
        float v = hval * wlog[t2];
        v += __shfl_xor(v, 1);
        v += __shfl_xor(v, 2);
        v += __shfl_xor(v, 4);
        v += __shfl_xor(v, 8);
        if (nl == 0) atomicAdd(&logits[(size_t)bg * TT + t], v);
      }
    }

    if (s != TT - 1) gbar(mybar, 48u * (unsigned)(s + 1));
  }
}

// ---------------------------------------------------------------------------
// Head: softmax over T, inclusive cumsum, rational Bezier eval. One block per b.
// b_log is shift-invariant under softmax and therefore omitted.
// ---------------------------------------------------------------------------
__global__ __launch_bounds__(TT)
void head_kernel(const float* __restrict__ logits,
                 const float* __restrict__ cpts,
                 const float* __restrict__ ratw,
                 float* __restrict__ out)
{
  __shared__ float sm[TT];
  __shared__ float sc[TT];
  __shared__ float stot;
  const int b = blockIdx.x;
  const int i = threadIdx.x;
  sm[i] = logits[(size_t)b * TT + i];
  __syncthreads();
  if (i == 0) {
    float mx = sm[0];
    for (int k = 1; k < TT; ++k) mx = fmaxf(mx, sm[k]);
    float run = 0.f;
    for (int k = 0; k < TT; ++k) { run += __expf(sm[k] - mx); sc[k] = run; }
    stot = run;
  }
  __syncthreads();
  const float t = sc[i] / stot;      // cumsum(softmax) in (0,1]
  const float u = 1.f - t;
  float tp[NPTS], up[NPTS];
  tp[0] = 1.f; up[0] = 1.f;
  #pragma unroll
  for (int k = 1; k < NPTS; ++k) { tp[k] = tp[k - 1] * t; up[k] = up[k - 1] * u; }
  const float C[NPTS] = {1.f, 7.f, 21.f, 35.f, 35.f, 21.f, 7.f, 1.f};
  float sw = 0.f, cx = 0.f, cy = 0.f;
  #pragma unroll
  for (int k = 0; k < NPTS; ++k) {
    const float w = C[k] * tp[k] * up[NPTS - 1 - k] * ratw[b * NPTS + k];
    sw += w;
    cx += w * cpts[b * 16 + 2 * k + 0];
    cy += w * cpts[b * 16 + 2 * k + 1];
  }
  const float inv = 1.f / sw;
  out[((size_t)b * TT + i) * 2 + 0] = cx * inv;
  out[((size_t)b * TT + i) * 2 + 1] = cy * inv;
}

// Regularizer: mean squared consecutive control-point distance over (B, NPTS-1)
__global__ void reg_kernel(const float* __restrict__ cpts, float* __restrict__ out)
{
  __shared__ float red[256];
  const int tid = threadIdx.x;
  float acc = 0.f;
  for (int i = tid; i < BB * (NPTS - 1); i += 256) {
    const int b = i / (NPTS - 1), s = i % (NPTS - 1);
    const float dx = cpts[b * 16 + (s + 1) * 2 + 0] - cpts[b * 16 + s * 2 + 0];
    const float dy = cpts[b * 16 + (s + 1) * 2 + 1] - cpts[b * 16 + s * 2 + 1];
    acc += dx * dx + dy * dy;
  }
  red[tid] = acc;
  __syncthreads();
  for (int o = 128; o > 0; o >>= 1) {
    if (tid < o) red[tid] += red[tid + o];
    __syncthreads();
  }
  if (tid == 0) out[(size_t)BB * TT * 2] = red[0] / (float)(BB * (NPTS - 1));
}

// ---------------------------------------------------------------------------
extern "C" void kernel_launch(void* const* d_in, const int* in_sizes, int n_in,
                              void* d_out, int out_size, void* d_ws, size_t ws_size,
                              hipStream_t stream)
{
  (void)in_sizes; (void)n_in; (void)out_size; (void)ws_size;
  const float* x     = (const float*)d_in[0];
  const float* cpts  = (const float*)d_in[1];
  const float* ratw  = (const float*)d_in[2];
  const float* W_hf  = (const float*)d_in[3];
  const float* b_hf  = (const float*)d_in[4];
  const float* W_hb  = (const float*)d_in[5];
  const float* b_hb  = (const float*)d_in[6];
  const float* W_cf  = (const float*)d_in[7];
  const float* b_cf  = (const float*)d_in[8];
  const float* W_cb  = (const float*)d_in[9];
  const float* b_cb  = (const float*)d_in[10];
  const float* Wih_f = (const float*)d_in[11];
  const float* Whh_f = (const float*)d_in[12];
  const float* bih_f = (const float*)d_in[13];
  const float* bhh_f = (const float*)d_in[14];
  const float* Wih_b = (const float*)d_in[15];
  const float* Whh_b = (const float*)d_in[16];
  const float* bih_b = (const float*)d_in[17];
  const float* bhh_b = (const float*)d_in[18];
  const float* Wlog  = (const float*)d_in[19];
  // d_in[20] = b_log: cancels under softmax, intentionally unused.

  char* ws = (char*)d_ws;                       // layout (256B aligned):
  unsigned* bar  = (unsigned*)(ws + 0);         // [2] barrier counters
  float*   logits = (float*)(ws + 256);         // [192][320]
  bf16_t*  hping  = (bf16_t*)(ws + 246016);     // [2 dir][2 phase][192][1024] bf16
  float*   cbuf   = (float*)(ws + 1818880);     // [2 dir][192][1024] f32
  bf16_t*  Wbf    = (bf16_t*)(ws + 3391744);    // [2 dir][4096][1024] bf16
  float*   out    = (float*)d_out;              // curves (192*320*2) + reg (1)

  setup_kernel<<<2048, 256, 0, stream>>>(cpts, ratw, W_hf, b_hf, W_hb, b_hb,
                                         W_cf, b_cf, W_cb, b_cb, Whh_f, Whh_b,
                                         Wbf, hping, cbuf, logits, bar);
  lstm_persistent_kernel<<<96, 256, 0, stream>>>(x, Wih_f, bih_f, bhh_f,
                                                 Wih_b, bih_b, bhh_b, Wlog,
                                                 Wbf, hping, cbuf, logits, bar);
  head_kernel<<<BB, TT, 0, stream>>>(logits, cpts, ratw, out);
  reg_kernel<<<1, 256, 0, stream>>>(cpts, out);
}
